// Decoder_55009941127205
// MI455X (gfx1250) — matrix operand order repaired and checked by reading, not compiled, not run on hardware
//
#include <hip/hip_runtime.h>
#include <hip/hip_bf16.h>
#include <stdint.h>

// ---------------------------------------------------------------------------
// LSTM decoder for MI455X (gfx1250, wave32, WMMA + Tensor Data Mover).
// B=64, L=64, I=128, H=1024, 8 autoregressive outer iterations.
// bf16 v_wmma_f32_16x16x32_bf16 GEMMs, fp32 cell state, fused per-timestep
// kernel. h is staged into LDS once per WG per step via TDM (tensor_load_to_lds)
// so the 4 gate-waves share A-fragments through ds_load_b128 instead of 4x
// redundant global traffic. Weight streams stay global (L2-resident, unique
// per WG).
// ---------------------------------------------------------------------------

typedef __attribute__((ext_vector_type(16))) __bf16 v16bf;
typedef __attribute__((ext_vector_type(8)))  float  v8f;
typedef __attribute__((ext_vector_type(4)))  unsigned int v4u;
typedef __attribute__((ext_vector_type(8)))  int    v8i_;
typedef __attribute__((ext_vector_type(4)))  int    v4i_;

#define B_BATCH 64
#define LSEQ    64
#define IDIM    128
#define HDIM    1024
#define SEQ_OUT 8

// ---- A-fragment loader (16-bit A 16x32, ISA 7.12.2) -----------------------
// lane = khalf*16 + m ; e=0..7  -> k = k0 + khalf*8 + e
//                       e=8..15 -> k = k0 + 16 + khalf*8 + (e-8)
union AFrag { uint4 u[2]; v16bf v; };

__device__ __forceinline__ v16bf load_a_frag(const __bf16* rowbase, int k0, int khalf) {
    AFrag a;
    const __bf16* p = rowbase + k0 + khalf * 8;
    a.u[0] = *(const uint4*)(p);
    a.u[1] = *(const uint4*)(p + 16);
    return a.v;
}

// ---- TDM: DMA `rows` x 1024 bf16 rows from global to LDS ------------------
// Descriptor per CDNA5 ISA §8.3/§8.4: 2D tile, data_size=2B,
// tile_dim0 = tensor_dim0 = tensor_dim0_stride = 1024, tile_dim1 = rows.
// This toolchain exposes the 6-arg tensor_load_to_lds builtin
// (g0, g1, g2, g3, extra, cpol); groups 2/3/extra are zero for a 2D tensor.
__device__ __forceinline__ void tdm_load_rows_to_lds(const __bf16* gsrc,
                                                     unsigned lds_off, int rows) {
    unsigned long long ga = (unsigned long long)(size_t)gsrc;
    v4u g0;
    g0.x = 1u;                                   // count=1 (valid descriptor)
    g0.y = lds_off;                              // lds_addr (bytes)
    g0.z = (unsigned)ga;                         // global_addr[31:0]
    g0.w = (unsigned)((ga >> 32) & 0x01ffffffu)  // global_addr[56:32]
         | (2u << 30);                           // type=2 ("image")
    v8i_ g1;
    g1[0] = 0x00010000;                          // data_size=1 -> 2 bytes
    g1[1] = (int)((unsigned)HDIM << 16);         // tensor_dim0 = 1024 (bits 79:48)
    g1[2] = (int)((unsigned)rows << 16);         // tensor_dim1 (bits 111:80)
    g1[3] = (int)((unsigned)HDIM << 16);         // tile_dim0 = 1024 (bits 127:112)
    g1[4] = rows;                                // tile_dim1 (bits 143:128)
    g1[5] = HDIM;                                // tensor_dim0_stride (bits 207:160)
    g1[6] = 0;
    g1[7] = 0;
    v4i_ z4 = {0, 0, 0, 0};                      // groups 2/3 unused (2D tensor)
    v8i_ z8 = {0, 0, 0, 0, 0, 0, 0, 0};
    __builtin_amdgcn_tensor_load_to_lds(g0, g1, z4, z4, z8, 0);
}

// ---- Weight pack: W[N,K] fp32 -> bf16 B-fragments -------------------------
// out[((nt*KT + kt)*32 + lane)*16 + e] = W[(nt*16 + (lane&15))*K + kt*32 + (e>>3)*16 + (lane>>4)*8 + (e&7)]
__global__ void pack_b_kernel(const float* __restrict__ W, __bf16* __restrict__ out,
                              int N, int K) {
    int idx = blockIdx.x * blockDim.x + threadIdx.x;
    if (idx >= N * K) return;
    int e    = idx & 15;
    int lane = (idx >> 4) & 31;
    int tile = idx >> 9;
    int KT   = K >> 5;
    int kt   = tile % KT;
    int nt   = tile / KT;
    int n = nt * 16 + (lane & 15);
    int k = kt * 32 + (e >> 3) * 16 + (lane >> 4) * 8 + (e & 7);
    out[idx] = (__bf16)W[(size_t)n * K + k];
}

__global__ void f32_to_bf16_kernel(const float* __restrict__ in, __bf16* __restrict__ out, int n) {
    int i = blockIdx.x * blockDim.x + threadIdx.x;
    if (i < n) out[i] = (__bf16)in[i];
}

__global__ void copy_f32_kernel(const float* __restrict__ in, float* __restrict__ out, int n) {
    int i = blockIdx.x * blockDim.x + threadIdx.x;
    if (i < n) out[i] = in[i];
}

// ---------------------------------------------------------------------------
// Fused LSTM timestep: gates = h@W_hh^T + x_t@W_ih^T + b ; cell update.
// grid = 64 workgroups (16 hidden columns each), block = 128 threads (4 waves).
// Wave g computes gate g's [64 x 16] tile: 4 M-tiles, 32+4 WMMA K-steps.
// h is TDM-staged into LDS (each wave DMAs its 16 rows).
// ---------------------------------------------------------------------------
__global__ void __launch_bounds__(128)
lstm_step_kernel(const __bf16* __restrict__ h_in,    // [64,1024]
                 __bf16* __restrict__ h_out,         // [64,1024]
                 float*  __restrict__ c,             // [64,1024]
                 const __bf16* __restrict__ x,       // [64,64,128]
                 const __bf16* __restrict__ whh_p,   // packed [256 nt][32 kt][32 lane][16]
                 const __bf16* __restrict__ wih_p,   // packed [256 nt][ 4 kt][32 lane][16]
                 const float* __restrict__ b_ih,     // [4096]
                 const float* __restrict__ b_hh,     // [4096]
                 __bf16* __restrict__ y,             // [64,64,1024]
                 int t)
{
    __shared__ __bf16 hsh[B_BATCH * HDIM];           // 128 KB staged h
    __shared__ float  glds[4 * B_BATCH * 16];        // 16 KB gate tiles [64x16] x4
    const int tid   = threadIdx.x;
    const int lane  = tid & 31;
    const int wave  = tid >> 5;          // gate index 0..3 (i,f,g,o)
    const int ns    = blockIdx.x;        // hidden column slice 0..63
    const int n0    = ns * 16;
    const int khalf = lane >> 4;
    const int lrow  = lane & 15;
    const int col_tile = wave * 64 + ns; // column tile within 4096-wide gates

    // ---- TDM-stage h: wave w DMAs rows [16w, 16w+16) into LDS ------------
    tdm_load_rows_to_lds(h_in + (size_t)wave * 16 * HDIM,
                         (unsigned)(size_t)&hsh[wave * 16 * HDIM], 16);
    __builtin_amdgcn_s_wait_tensorcnt(0);
    __syncthreads();

    v8f acc[4] = {};                     // 4 M-tiles x 8 f32

    // ---- h @ W_hh^T : K = 1024 -> 32 WMMA steps (A from LDS) -------------
    const __bf16* bbase = whh_p + (size_t)col_tile * 32 * 512;
    for (int kt = 0; kt < 32; ++kt) {
        v16bf bf = *(const v16bf*)(bbase + (size_t)kt * 512 + lane * 16);
#pragma unroll
        for (int m = 0; m < 4; ++m) {
            const __bf16* arow = hsh + (size_t)(m * 16 + lrow) * HDIM;
            v16bf af = load_a_frag(arow, kt * 32, khalf);
            acc[m] = __builtin_amdgcn_wmma_f32_16x16x32_bf16(
                         false, af, false, bf, (short)0, acc[m], false, false);
        }
    }

    // ---- x_t @ W_ih^T : K = 128 -> 4 WMMA steps (A from global) ----------
    const __bf16* bbase2 = wih_p + (size_t)col_tile * 4 * 512;
    for (int kt = 0; kt < 4; ++kt) {
        v16bf bf = *(const v16bf*)(bbase2 + (size_t)kt * 512 + lane * 16);
#pragma unroll
        for (int m = 0; m < 4; ++m) {
            const __bf16* arow = x + ((size_t)(m * 16 + lrow) * LSEQ + t) * IDIM;
            v16bf af = load_a_frag(arow, kt * 32, khalf);
            acc[m] = __builtin_amdgcn_wmma_f32_16x16x32_bf16(
                         false, af, false, bf, (short)0, acc[m], false, false);
        }
    }

    // ---- stash gate tiles to LDS (C/D layout: row = 16m + 8*khalf + r) ---
    float* gl = glds + wave * (B_BATCH * 16);
#pragma unroll
    for (int m = 0; m < 4; ++m)
#pragma unroll
        for (int r = 0; r < 8; ++r)
            gl[(m * 16 + khalf * 8 + r) * 16 + lrow] = acc[m][r];
    __syncthreads();

    // ---- gate activations + cell/hidden update ---------------------------
    for (int e = tid; e < B_BATCH * 16; e += 128) {
        int row = e >> 4;                // batch index
        int col = e & 15;
        int n   = n0 + col;
        float vi = glds[0 * 1024 + e] + b_ih[n]            + b_hh[n];
        float vf = glds[1 * 1024 + e] + b_ih[HDIM + n]     + b_hh[HDIM + n];
        float vg = glds[2 * 1024 + e] + b_ih[2 * HDIM + n] + b_hh[2 * HDIM + n];
        float vo = glds[3 * 1024 + e] + b_ih[3 * HDIM + n] + b_hh[3 * HDIM + n];
        float ig = 1.0f / (1.0f + __expf(-vi));
        float fg = 1.0f / (1.0f + __expf(-vf));
        float gg = tanhf(vg);
        float og = 1.0f / (1.0f + __expf(-vo));
        size_t ci = (size_t)row * HDIM + n;
        float cn = fg * c[ci] + ig * gg;
        c[ci] = cn;
        float hn = og * tanhf(cn);
        __bf16 hb = (__bf16)hn;
        h_out[ci] = hb;                                   // recurrent carry
        y[((size_t)row * LSEQ + t) * HDIM + n] = hb;      // sequence output
    }
}

// ---------------------------------------------------------------------------
// FC: out = y @ W_fc^T + b_fc. [4096,1024] x [1024,128].
// grid = 256 M-tiles, 4 waves; wave w owns 2 N-tiles. y rows TDM-staged into
// LDS (shared by all waves). Writes next x (bf16) and channel 127 into d_out.
// ---------------------------------------------------------------------------
__global__ void __launch_bounds__(128)
fc_kernel(const __bf16* __restrict__ y,       // [4096,1024]
          const __bf16* __restrict__ wfc_p,   // packed [8 nt][32 kt][32 lane][16]
          const float* __restrict__ b_fc,     // [128]
          __bf16* __restrict__ x_next,        // [4096,128]
          float* __restrict__ out,            // [64, 512]
          int it)
{
    __shared__ __bf16 ysh[16 * HDIM];         // 32 KB: this block's 16 A-rows
    const int tid   = threadIdx.x;
    const int lane  = tid & 31;
    const int wave  = tid >> 5;
    const int khalf = lane >> 4;
    const int lrow  = lane & 15;
    const int m0    = blockIdx.x * 16;

    if (wave == 0)
        tdm_load_rows_to_lds(y + (size_t)m0 * HDIM, (unsigned)(size_t)&ysh[0], 16);
    __builtin_amdgcn_s_wait_tensorcnt(0);     // no-op for waves with cnt==0
    __syncthreads();

    v8f acc[2] = {};
    for (int kt = 0; kt < 32; ++kt) {
        const __bf16* arow = ysh + (size_t)lrow * HDIM;
        v16bf af = load_a_frag(arow, kt * 32, khalf);
#pragma unroll
        for (int j = 0; j < 2; ++j) {
            int ct = wave * 2 + j;
            v16bf bf = *(const v16bf*)(wfc_p + ((size_t)ct * 32 + kt) * 512 + lane * 16);
            acc[j] = __builtin_amdgcn_wmma_f32_16x16x32_bf16(
                         false, af, false, bf, (short)0, acc[j], false, false);
        }
    }

#pragma unroll
    for (int j = 0; j < 2; ++j) {
        int cc = (wave * 2 + j) * 16 + lrow;
        float bias = b_fc[cc];
#pragma unroll
        for (int r = 0; r < 8; ++r) {
            int rr = m0 + khalf * 8 + r;         // row = b*64 + l
            float v = acc[j][r] + bias;
            x_next[(size_t)rr * IDIM + cc] = (__bf16)v;
            if (cc == IDIM - 1) {
                int b = rr >> 6, l = rr & 63;
                out[(size_t)b * (LSEQ * SEQ_OUT) + it * LSEQ + l] = v;
            }
        }
    }
}

// ---------------------------------------------------------------------------
extern "C" void kernel_launch(void* const* d_in, const int* in_sizes, int n_in,
                              void* d_out, int out_size, void* d_ws, size_t ws_size,
                              hipStream_t stream) {
    const float* x_f  = (const float*)d_in[0];   // [64,64,128]
    const float* hid  = (const float*)d_in[1];   // [1,64,1024]
    const float* cel  = (const float*)d_in[2];   // [1,64,1024]
    const float* W_ih = (const float*)d_in[3];   // [4096,128]
    const float* W_hh = (const float*)d_in[4];   // [4096,1024]
    const float* b_ih = (const float*)d_in[5];   // [4096]
    const float* b_hh = (const float*)d_in[6];   // [4096]
    const float* W_fc = (const float*)d_in[7];   // [128,1024]
    const float* b_fc = (const float*)d_in[8];   // [128]
    float* out = (float*)d_out;

    // workspace carve-up (~20 MB), 256-byte aligned
    char* ws = (char*)d_ws;
    size_t off = 0;
    auto carve = [&](size_t bytes) {
        char* p = ws + off;
        off = (off + bytes + 255) & ~(size_t)255;
        return p;
    };
    __bf16* whh_p = (__bf16*)carve((size_t)4096 * 1024 * 2);
    __bf16* wih_p = (__bf16*)carve((size_t)4096 * 128 * 2);
    __bf16* wfc_p = (__bf16*)carve((size_t)128 * 1024 * 2);
    __bf16* hbuf0 = (__bf16*)carve((size_t)64 * 1024 * 2);
    __bf16* hbuf1 = (__bf16*)carve((size_t)64 * 1024 * 2);
    float*  cbuf  = (float*)carve((size_t)64 * 1024 * 4);
    __bf16* xbuf  = (__bf16*)carve((size_t)64 * 64 * 128 * 2);
    __bf16* ybuf  = (__bf16*)carve((size_t)64 * 64 * 1024 * 2);
    __bf16* hbufs[2] = {hbuf0, hbuf1};

    // per-call packs + state init — deterministic
    pack_b_kernel<<<(4096 * 1024) / 256, 256, 0, stream>>>(W_hh, whh_p, 4096, 1024);
    pack_b_kernel<<<(4096 * 128) / 256, 256, 0, stream>>>(W_ih, wih_p, 4096, 128);
    pack_b_kernel<<<(128 * 1024) / 256, 256, 0, stream>>>(W_fc, wfc_p, 128, 1024);
    f32_to_bf16_kernel<<<(64 * 1024) / 256, 256, 0, stream>>>(hid, hbuf0, 64 * 1024);
    copy_f32_kernel<<<(64 * 1024) / 256, 256, 0, stream>>>(cel, cbuf, 64 * 1024);
    f32_to_bf16_kernel<<<(64 * 64 * 128) / 256, 256, 0, stream>>>(x_f, xbuf, 64 * 64 * 128);

    int p = 0;
    for (int it = 0; it < SEQ_OUT; ++it) {
        for (int t = 0; t < LSEQ; ++t) {
            lstm_step_kernel<<<64, 128, 0, stream>>>(
                hbufs[p], hbufs[p ^ 1], cbuf, xbuf, whh_p, wih_p,
                b_ih, b_hh, ybuf, t);
            p ^= 1;
        }
        // fc reads ybuf, overwrites xbuf (consumed) for the next iteration
        fc_kernel<<<256, 128, 0, stream>>>(ybuf, wfc_p, b_fc, xbuf, out, it);
    }
}
